// Spatial_RNN_66340064854377
// MI455X (gfx1250) — compile-verified, matrix operand order checked by reading, not run on hardware
//
#include <hip/hip_runtime.h>
#include <hip/hip_bf16.h>

typedef __attribute__((ext_vector_type(16))) _Float16 v16h;
typedef __attribute__((ext_vector_type(8)))  _Float16 v8h;
typedef __attribute__((ext_vector_type(8)))  float    v8f;

#define BATCH 8

// ---------------------------------------------------------------------------
// Implicit-GEMM conv, stride 1, square kernel, zero padding.
//   M = 16 output pixels (one row, 16 consecutive cols)
//   N = 16 output channels
//   K = ntaps * Cin_pad32, consumed 32 channels/tap at a time via
//       v_wmma_f32_16x16x32_f16.
// One wave (32 threads) per tile. act: 0=none 1=relu 2=tanh 3=sigmoid
//
// Fragment layout (16-bit A 16x32, ISA 7.12.2): lane L holds row M=L%16,
// half h=L/16; element e -> local K = e + 8h + (e>=8 ? 8 : 0), i.e. two
// contiguous 8-element runs at {8h, 16+8h}. We lay out both LDS tiles and
// weights with cin fastest so each run is one 16-byte load (ds_load_b128 /
// global_load_b128).
//
// LDS staging: slab [ky][x][cin_pad] of f16, x in [0, 16+KW-1), filled with
// x-coalesced global loads. Max size 3*18*96 = 5184 halves (~10.1 KB).
// ---------------------------------------------------------------------------
__global__ __launch_bounds__(32) void conv_wmma(
    const float* __restrict__ in, const _Float16* __restrict__ wf,
    const float* __restrict__ bias, float* __restrict__ out,
    int Cin, int Cout, int H, int W, int KH, int KW, int pad, int act)
{
    __shared__ __align__(16) _Float16 smem[5504];

    const int TW    = 16 + KW - 1;
    const int Cinp  = (Cin + 31) & ~31;
    const int ntaps = KH * KW;

    const int tilesX = W >> 4;
    const int oy = blockIdx.x / tilesX;
    const int x0 = (blockIdx.x % tilesX) << 4;
    const int n0 = blockIdx.y << 4;
    const int b  = blockIdx.z;

    const int lane = threadIdx.x;
    const int mrow = lane & 15;   // A: pixel-in-tile ; B/C: cout-in-tile
    const int half = lane >> 4;

    const long HW = (long)H * W;
    const float* inB = in + (long)b * Cin * HW;

    // ---- stage input slab into LDS (f16), x-coalesced global reads ----
    const int slab = TW * Cinp;
    for (int ky = 0; ky < KH; ++ky) {
        const int iy = oy + ky - pad;
        for (int idx = lane; idx < slab; idx += 32) {
            const int x   = idx % TW;
            const int cin = idx / TW;
            const int ix  = x0 + x - pad;
            _Float16 v = (_Float16)0.f;
            if (cin < Cin && iy >= 0 && iy < H && ix >= 0 && ix < W)
                v = (_Float16)inB[cin * HW + (long)iy * W + ix];
            smem[(ky * TW + x) * Cinp + cin] = v;
        }
    }
    __syncthreads();

    // ---- WMMA main loop: 2 ds_load_b128 + 2 global_load_b128 + 1 wmma ----
    const int cout = n0 + mrow;                      // weights are Cout-padded
    const _Float16* wrow = wf + (long)cout * ntaps * Cinp + 8 * half;

    v8f accv = {};
    for (int ky = 0; ky < KH; ++ky) {
        for (int kx = 0; kx < KW; ++kx) {
            const _Float16* arow = smem + (ky * TW + mrow + kx) * Cinp + 8 * half;
            const _Float16* brow = wrow + (ky * KW + kx) * Cinp;
            for (int cb = 0; cb < Cinp; cb += 32) {
                v8h alo = *(const v8h*)(arow + cb);
                v8h ahi = *(const v8h*)(arow + cb + 16);
                v8h blo = *(const v8h*)(brow + cb);
                v8h bhi = *(const v8h*)(brow + cb + 16);
                v16h afrag = __builtin_shufflevector(alo, ahi,
                    0,1,2,3,4,5,6,7,8,9,10,11,12,13,14,15);
                v16h bfrag = __builtin_shufflevector(blo, bhi,
                    0,1,2,3,4,5,6,7,8,9,10,11,12,13,14,15);
                accv = __builtin_amdgcn_wmma_f32_16x16x32_f16(
                    false, afrag, false, bfrag, (short)0, accv, false, false);
            }
        }
    }

    // ---- epilogue: bias + activation, store f32 ----
    if (cout < Cout) {
        const float bb = bias[cout];
        float* outB = out + ((long)b * Cout + cout) * HW + (long)oy * W + x0;
#pragma unroll
        for (int j = 0; j < 8; ++j) {
            float v = accv[j] + bb;
            if (act == 1)      v = fmaxf(v, 0.f);
            else if (act == 2) v = tanhf(v);
            else if (act == 3) v = 1.f / (1.f + expf(-v));
            outB[j + 8 * half] = v;
        }
    }
}

// ---------------------------------------------------------------------------
// Weight repack: OIHW f32 -> [Cout_pad][tap][Cin_pad] f16, zero padded.
__global__ __launch_bounds__(256) void cvt_w_k(
    const float* __restrict__ src, _Float16* __restrict__ dst,
    int Cout, int Cin, int ntaps, int Coutp, int Cinp)
{
    int n = Coutp * ntaps * Cinp;
    int i = blockIdx.x * blockDim.x + threadIdx.x;
    if (i >= n) return;
    int cin = i % Cinp; int t = i / Cinp;
    int tap = t % ntaps; int cout = t / ntaps;
    _Float16 v = (_Float16)0.f;
    if (cout < Cout && cin < Cin)
        v = (_Float16)src[((long)cout * Cin + cin) * ntaps + tap];
    dst[i] = v;
}

__global__ __launch_bounds__(256) void pool2_k(
    const float* __restrict__ in, float* __restrict__ out, int BC, int Ho, int Wo)
{
    int i = blockIdx.x * blockDim.x + threadIdx.x;
    int total = BC * Ho * Wo;
    if (i >= total) return;
    int ox = i % Wo; int t = i / Wo;
    int oy = t % Ho; int bc = t / Ho;
    int Wi = Wo * 2;
    const float* p = in + (((long)bc * (Ho * 2) + oy * 2) * Wi) + ox * 2;
    out[i] = fmaxf(fmaxf(p[0], p[1]), fmaxf(p[Wi], p[Wi + 1]));
}

// bilinear x2, half-pixel centers (align_corners=False); writes into concat
// buffer with per-image channel capacity Ccat at channel offset 0.
__global__ __launch_bounds__(256) void upsample2_k(
    const float* __restrict__ in, float* __restrict__ cat,
    int C, int Hi, int Wi, int Ccat)
{
    int Ho = Hi * 2, Wo = Wi * 2;
    int i = blockIdx.x * blockDim.x + threadIdx.x;
    int total = BATCH * C * Ho * Wo;
    if (i >= total) return;
    int ox = i % Wo; int t = i / Wo;
    int oy = t % Ho; t /= Ho;
    int c  = t % C;  int b = t / C;

    float sy = (oy + 0.5f) * 0.5f - 0.5f;
    float sx = (ox + 0.5f) * 0.5f - 0.5f;
    int y0 = (int)floorf(sy); float wy = sy - (float)y0;
    int xx0 = (int)floorf(sx); float wx = sx - (float)xx0;
    int y0c = min(max(y0, 0), Hi - 1);
    int y1c = min(max(y0 + 1, 0), Hi - 1);
    int x0c = min(max(xx0, 0), Wi - 1);
    int x1c = min(max(xx0 + 1, 0), Wi - 1);

    const float* p = in + ((long)(b * C + c) * Hi) * Wi;
    float v00 = p[(long)y0c * Wi + x0c], v01 = p[(long)y0c * Wi + x1c];
    float v10 = p[(long)y1c * Wi + x0c], v11 = p[(long)y1c * Wi + x1c];
    float v = (1.f - wy) * ((1.f - wx) * v00 + wx * v01)
            +        wy  * ((1.f - wx) * v10 + wx * v11);
    cat[((long)(b * Ccat + c) * Ho + oy) * Wo + ox] = v;
}

__global__ __launch_bounds__(256) void copy_skip_k(
    const float* __restrict__ in, float* __restrict__ cat,
    int C, int H, int W, int Ccat, int coff)
{
    int i = blockIdx.x * blockDim.x + threadIdx.x;
    int total = BATCH * C * H * W;
    if (i >= total) return;
    int px = i % (H * W); int t = i / (H * W);
    int c = t % C; int b = t / C;
    cat[((long)(b * Ccat + coff + c)) * H * W + px] = in[i];
}

__global__ __launch_bounds__(256) void fill_k(float* __restrict__ p, int n, float v)
{
    int i = blockIdx.x * blockDim.x + threadIdx.x;
    if (i < n) p[i] = v;
}

// ---------------------------------------------------------------------------
// Directional linear recurrence: h_k = (1-p)*x + p*h_{k-1}.
// axis==3: scan along W (thread owns one row); axis==2: scan along H
// (thread owns one column). flip: read spatially reversed, write in SCAN
// order (the reference keeps scan order for reverse directions).
// acc = max(acc, h) elementwise; each thread owns a disjoint line.
// ---------------------------------------------------------------------------
__global__ __launch_bounds__(256) void lrnn_scan_k(
    const float* __restrict__ X, const float* __restrict__ FM,
    float* __restrict__ acc, int H, int W, int c_off, int axis, int flip)
{
    int lines = (axis == 3) ? H : W;
    int tid = blockIdx.x * blockDim.x + threadIdx.x;
    int total = BATCH * 16 * lines;
    if (tid >= total) return;
    int line = tid % lines; int t = tid / lines;
    int ch = t % 16; int b = t / 16;
    int L = (axis == 3) ? W : H;

    const long basex = (long)(b * 16 + ch) * H * W;
    const long basep = (long)(b * 64 + c_off + ch) * H * W;

    float h = 0.f;
    for (int k = 0; k < L; ++k) {
        int pos = flip ? (L - 1 - k) : k;
        long roff = (axis == 3) ? ((long)line * W + pos) : ((long)pos * W + line);
        if (k + 8 < L) {
            int pp = flip ? (L - 9 - k) : (k + 8);
            long poff = (axis == 3) ? ((long)line * W + pp) : ((long)pp * W + line);
            __builtin_prefetch(&X[basex + poff], 0, 1);
            __builtin_prefetch(&FM[basep + poff], 0, 1);
        }
        float x = X[basex + roff];
        float p = FM[basep + roff];
        h = (1.f - p) * x + p * h;
        long woff = (axis == 3) ? ((long)line * W + k) : ((long)k * W + line);
        float old = acc[basex + woff];
        acc[basex + woff] = fmaxf(old, h);
    }
}

// ---------------------------------------------------------------------------
extern "C" void kernel_launch(void* const* d_in, const int* in_sizes, int n_in,
                              void* d_out, int out_size, void* d_ws, size_t ws_size,
                              hipStream_t stream)
{
    const float* img = (const float*)d_in[0];
    const float* w1 = (const float*)d_in[1];  const float* b1 = (const float*)d_in[2];
    const float* w2 = (const float*)d_in[3];  const float* b2 = (const float*)d_in[4];
    const float* w3 = (const float*)d_in[5];  const float* b3 = (const float*)d_in[6];
    const float* w4 = (const float*)d_in[7];  const float* b4 = (const float*)d_in[8];
    const float* w5 = (const float*)d_in[9];  const float* b5 = (const float*)d_in[10];
    const float* w6 = (const float*)d_in[11]; const float* b6 = (const float*)d_in[12];
    const float* w7 = (const float*)d_in[13]; const float* b7 = (const float*)d_in[14];
    const float* w8 = (const float*)d_in[15]; const float* b8 = (const float*)d_in[16];
    const float* w9 = (const float*)d_in[17]; const float* b9 = (const float*)d_in[18];
    const float* wi = (const float*)d_in[19]; const float* bi = (const float*)d_in[20];
    const float* wo = (const float*)d_in[21]; const float* bo = (const float*)d_in[22];

    char* base = (char*)d_ws;
    size_t off = 0;
    auto alloc = [&](size_t nbytes) -> char* {
        char* p = base + off;
        off += (nbytes + 255) & ~(size_t)255;
        return p;
    };

    // fp32 activation temps
    float* x1   = (float*)alloc(8388608ull  * 4);   // 8x16x256x256
    float* p1   = (float*)alloc(2097152ull  * 4);   // 8x16x128x128
    float* x2   = (float*)alloc(4194304ull  * 4);   // 8x32x128x128
    float* p2   = (float*)alloc(1048576ull  * 4);   // 8x32x64x64
    float* x3   = (float*)alloc(1048576ull  * 4);
    float* p3   = (float*)alloc(262144ull   * 4);   // 8x32x32x32
    float* x4   = (float*)alloc(262144ull   * 4);
    float* p4   = (float*)alloc(65536ull    * 4);   // 8x32x16x16
    float* x5   = (float*)alloc(131072ull   * 4);   // 8x64x16x16
    float* cat6 = (float*)alloc(786432ull   * 4);   // 8x96x32x32
    float* x6   = (float*)alloc(262144ull   * 4);
    float* cat7 = (float*)alloc(2097152ull  * 4);   // 8x64x64x64
    float* x7   = (float*)alloc(1048576ull  * 4);
    float* cat8 = (float*)alloc(8388608ull  * 4);   // 8x64x128x128
    float* x8   = (float*)alloc(4194304ull  * 4);   // 8x32x128x128
    float* cat9 = (float*)alloc(25165824ull * 4);   // 8x48x256x256
    float* fm   = (float*)alloc(33554432ull * 4);   // 8x64x256x256
    float* xin  = cat8;   // alias: cat8 dead after conv8
    float* accb = cat9;   // alias: cat9 dead after conv9

    // f16 weight buffers [Cout_pad16][tap][Cin_pad32]
    _Float16* wf1 = (_Float16*)alloc(12800 * 2);  // 16*25*32
    _Float16* wf2 = (_Float16*)alloc(9216  * 2);  // 32*9*32
    _Float16* wf3 = (_Float16*)alloc(9216  * 2);
    _Float16* wf4 = (_Float16*)alloc(9216  * 2);
    _Float16* wf5 = (_Float16*)alloc(18432 * 2);  // 64*9*32
    _Float16* wf6 = (_Float16*)alloc(27648 * 2);  // 32*9*96
    _Float16* wf7 = (_Float16*)alloc(18432 * 2);  // 32*9*64
    _Float16* wf8 = (_Float16*)alloc(18432 * 2);
    _Float16* wf9 = (_Float16*)alloc(36864 * 2);  // 64*9*64
    _Float16* wfi = (_Float16*)alloc(4608  * 2);  // 16*9*32
    _Float16* wfo = (_Float16*)alloc(4608  * 2);  // 16(pad from 3)*9*32

    auto cvtw = [&](const float* src, _Float16* dst, int Cout, int Cin, int ntaps) {
        int Coutp = (Cout + 15) & ~15, Cinp = (Cin + 31) & ~31;
        int n = Coutp * ntaps * Cinp;
        cvt_w_k<<<(n + 255) / 256, 256, 0, stream>>>(src, dst, Cout, Cin, ntaps,
                                                     Coutp, Cinp);
    };
    cvtw(w1, wf1, 16, 15, 25);
    cvtw(w2, wf2, 32, 16, 9);
    cvtw(w3, wf3, 32, 32, 9);
    cvtw(w4, wf4, 32, 32, 9);
    cvtw(w5, wf5, 64, 32, 9);
    cvtw(w6, wf6, 32, 96, 9);
    cvtw(w7, wf7, 32, 64, 9);
    cvtw(w8, wf8, 32, 64, 9);
    cvtw(w9, wf9, 64, 48, 9);
    cvtw(wi, wfi, 16, 15, 9);
    cvtw(wo, wfo, 3, 16, 9);

    auto conv = [&](const float* in, const _Float16* wf, const float* bias,
                    float* out, int Cin, int Cout, int HW, int KH, int pad, int act) {
        dim3 grid(HW * (HW >> 4), (Cout + 15) >> 4, BATCH);
        conv_wmma<<<grid, 32, 0, stream>>>(in, wf, bias, out,
                                           Cin, Cout, HW, HW, KH, KH, pad, act);
    };
    auto pool = [&](const float* in, float* out, int C, int Ho) {
        int n = BATCH * C * Ho * Ho;
        pool2_k<<<(n + 255) / 256, 256, 0, stream>>>(in, out, BATCH * C, Ho, Ho);
    };
    auto ups = [&](const float* in, float* cat, int C, int Hi, int Ccat) {
        int n = BATCH * C * Hi * 2 * Hi * 2;
        upsample2_k<<<(n + 255) / 256, 256, 0, stream>>>(in, cat, C, Hi, Hi, Ccat);
    };
    auto skip = [&](const float* in, float* cat, int C, int H, int Ccat, int coff) {
        int n = BATCH * C * H * H;
        copy_skip_k<<<(n + 255) / 256, 256, 0, stream>>>(in, cat, C, H, H, Ccat, coff);
    };

    // ---- encoder ----
    conv(img, wf1, b1, x1, 15, 16, 256, 5, 2, 1);
    pool(x1, p1, 16, 128);
    conv(p1, wf2, b2, x2, 16, 32, 128, 3, 1, 1);
    pool(x2, p2, 32, 64);
    conv(p2, wf3, b3, x3, 32, 32, 64, 3, 1, 1);
    pool(x3, p3, 32, 32);
    conv(p3, wf4, b4, x4, 32, 32, 32, 3, 1, 1);
    pool(x4, p4, 32, 16);
    conv(p4, wf5, b5, x5, 32, 64, 16, 3, 1, 1);

    // ---- decoder (upsample -> concat buffer -> conv) ----
    ups(x5, cat6, 64, 16, 96);  skip(x4, cat6, 32, 32, 96, 64);
    conv(cat6, wf6, b6, x6, 96, 32, 32, 3, 1, 1);
    ups(x6, cat7, 32, 32, 64);  skip(x3, cat7, 32, 64, 64, 32);
    conv(cat7, wf7, b7, x7, 64, 32, 64, 3, 1, 1);
    ups(x7, cat8, 32, 64, 64);  skip(x2, cat8, 32, 128, 64, 32);
    conv(cat8, wf8, b8, x8, 64, 32, 128, 3, 1, 1);
    ups(x8, cat9, 32, 128, 48); skip(x1, cat9, 16, 256, 48, 32);
    conv(cat9, wf9, b9, fm, 48, 64, 256, 3, 1, 2);   // tanh -> fm

    // ---- input feature conv (xin aliases cat8, dead by now) ----
    conv(img, wfi, bi, xin, 15, 16, 256, 3, 1, 0);

    // ---- 8-direction linear recurrence + running max (accb aliases cat9) ----
    {
        int n = 8388608;
        fill_k<<<(n + 255) / 256, 256, 0, stream>>>(accb, n, -3.0e38f);
        int ln = BATCH * 16 * 256;
        int blocks = (ln + 255) / 256;
        const int dirs[8][3] = {{0,3,0},{0,3,1},{16,2,0},{16,2,1},
                                {32,3,0},{32,3,1},{48,2,0},{48,2,1}};
        for (int i = 0; i < 8; ++i)
            lrnn_scan_k<<<blocks, 256, 0, stream>>>(xin, fm, accb, 256, 256,
                                                    dirs[i][0], dirs[i][1], dirs[i][2]);
    }

    // ---- output conv + sigmoid ----
    conv(accb, wfo, bo, (float*)d_out, 16, 3, 256, 3, 1, 3);
}
